// Attention_65309272703410
// MI455X (gfx1250) — matrix operand order repaired and checked by reading, not compiled, not run on hardware
//
#include <hip/hip_runtime.h>
#include <hip/hip_bf16.h>

typedef __attribute__((ext_vector_type(16))) _Float16 v16h;
typedef __attribute__((ext_vector_type(8)))  _Float16 v8h;
typedef __attribute__((ext_vector_type(8)))  float    v8f;

#define C_DIM 384
#define HEADS 6
#define HD    64
#define B_SZ  8
#define HH    56
#define WW    56
#define T1    3136        // 56*56
#define T2    784         // 28*28
#define T2P   800         // padded kv length (multiple of 32)
#define BN_EPS 1e-5f
#define SCALE  0.0510310363f   // 384^-0.5

// ---------------------------------------------------------------------------
// Weight convert: W[k][n] (f32, [in,out]) -> Wt[n][k] (f16) for 4 matrices
// ---------------------------------------------------------------------------
__global__ void cvt_weights(const float* __restrict__ Wq, const float* __restrict__ Wk,
                            const float* __restrict__ Wv, const float* __restrict__ Wp,
                            _Float16* __restrict__ out) {
    int idx = blockIdx.x * blockDim.x + threadIdx.x;
    if (idx >= 4 * C_DIM * C_DIM) return;
    int which = idx / (C_DIM * C_DIM);
    int r = idx % (C_DIM * C_DIM);
    int n = r / C_DIM;          // out channel
    int k = r % C_DIM;          // in channel
    const float* W = (which == 0) ? Wq : (which == 1) ? Wk : (which == 2) ? Wv : Wp;
    out[idx] = (_Float16)W[k * C_DIM + n];
}

// ---------------------------------------------------------------------------
// Depthwise 3x3 conv (stride 1, pad 1) + BN(eval) -> f16 tokens [B][T1][C]
// ---------------------------------------------------------------------------
__global__ void conv_bn_q(const float* __restrict__ x, const float* __restrict__ w,
                          const float* __restrict__ bn, _Float16* __restrict__ out) {
    size_t idx = (size_t)blockIdx.x * blockDim.x + threadIdx.x;
    if (idx >= (size_t)B_SZ * T1 * C_DIM) return;
    int c = (int)(idx % C_DIM);
    int t = (int)((idx / C_DIM) % T1);
    int b = (int)(idx / ((size_t)C_DIM * T1));
    int y = t / WW, xp = t % WW;
    float acc = 0.f;
    #pragma unroll
    for (int ky = 0; ky < 3; ky++) {
        int yy = y + ky - 1;
        if (yy < 0 || yy >= HH) continue;
        #pragma unroll
        for (int kx = 0; kx < 3; kx++) {
            int xx = xp + kx - 1;
            if (xx < 0 || xx >= WW) continue;
            acc += x[((size_t)b * T1 + yy * WW + xx) * C_DIM + c] * w[c * 9 + ky * 3 + kx];
        }
    }
    float g = bn[c], be = bn[C_DIM + c], m = bn[2 * C_DIM + c], va = bn[3 * C_DIM + c];
    float o = (acc - m) * (g * rsqrtf(va + BN_EPS)) + be;
    out[idx] = (_Float16)o;
}

// ---------------------------------------------------------------------------
// Depthwise 3x3 conv (stride 2, pad 1) + BN for k and v; pads rows [784,800)=0
// outputs [B][T2P][C] f16
// ---------------------------------------------------------------------------
__global__ void conv_bn_kv(const float* __restrict__ x,
                           const float* __restrict__ wk, const float* __restrict__ wv,
                           const float* __restrict__ bnk, const float* __restrict__ bnv,
                           _Float16* __restrict__ kout, _Float16* __restrict__ vout) {
    size_t idx = (size_t)blockIdx.x * blockDim.x + threadIdx.x;
    if (idx >= (size_t)B_SZ * T2P * C_DIM) return;
    int c = (int)(idx % C_DIM);
    int t = (int)((idx / C_DIM) % T2P);
    int b = (int)(idx / ((size_t)C_DIM * T2P));
    if (t >= T2) { kout[idx] = (_Float16)0.f; vout[idx] = (_Float16)0.f; return; }
    int y = t / 28, xp = t % 28;
    float ak = 0.f, av = 0.f;
    #pragma unroll
    for (int ky = 0; ky < 3; ky++) {
        int yy = 2 * y + ky - 1;
        if (yy < 0 || yy >= HH) continue;
        #pragma unroll
        for (int kx = 0; kx < 3; kx++) {
            int xx = 2 * xp + kx - 1;
            if (xx < 0 || xx >= WW) continue;
            float xv = x[((size_t)b * T1 + yy * WW + xx) * C_DIM + c];
            ak += xv * wk[c * 9 + ky * 3 + kx];
            av += xv * wv[c * 9 + ky * 3 + kx];
        }
    }
    {
        float g = bnk[c], be = bnk[C_DIM + c], m = bnk[2 * C_DIM + c], va = bnk[3 * C_DIM + c];
        kout[idx] = (_Float16)((ak - m) * (g * rsqrtf(va + BN_EPS)) + be);
    }
    {
        float g = bnv[c], be = bnv[C_DIM + c], m = bnv[2 * C_DIM + c], va = bnv[3 * C_DIM + c];
        vout[idx] = (_Float16)((av - m) * (g * rsqrtf(va + BN_EPS)) + be);
    }
}

// ---------------------------------------------------------------------------
// WMMA GEMM: Y = X[M x 384] @ W (Wt stored [N=384][K=384] f16).
// Block = 128 threads = 4 waves; each WAVE computes a 16x64 strip
// (4 accumulators) so the A fragment is loaded once per k-step and
// reused across 4 WMMAs (4x less A traffic than 16x16 tiles).
// grid.x = M/64 (4 m-tiles per block, one per wave), grid.y = N/64 = 6.
// mode: 0=f16 row-major out, 1=f16 transposed out (vT: [B][C][T2P]),
//       2=f32 out + bias.
// ---------------------------------------------------------------------------
__global__ __launch_bounds__(128) void gemm_wmma(const _Float16* __restrict__ X,
                                                 const _Float16* __restrict__ Wt,
                                                 const float* __restrict__ bias,
                                                 void* __restrict__ Yv, int mode) {
    int lane = threadIdx.x & 31, wave = threadIdx.x >> 5;
    int mt = blockIdx.x * 4 + wave;           // 16-row tile index for this wave
    int nbase = blockIdx.y * 64;              // 64 output cols per wave
    int half = lane >> 4, lm = lane & 15;
    int c0 = half * 16;
    const _Float16* xr = X + (size_t)(mt * 16 + lm) * C_DIM;
    const _Float16* wr = Wt + (size_t)(nbase + lm) * C_DIM + half * 16;
    const size_t wstep = (size_t)16 * C_DIM;  // 16 weight rows
    v8f acc0 = {}, acc1 = {}, acc2 = {}, acc3 = {};
    for (int kk = 0; kk < C_DIM; kk += 32) {
        v8h alo = *(const v8h*)(xr + kk + c0);
        v8h ahi = *(const v8h*)(xr + kk + c0 + 8);
        v16h a = __builtin_shufflevector(alo, ahi, 0,1,2,3,4,5,6,7,8,9,10,11,12,13,14,15);
        v16h b0 = *(const v16h*)(wr + kk);
        v16h b1 = *(const v16h*)(wr + wstep + kk);
        v16h b2 = *(const v16h*)(wr + 2 * wstep + kk);
        v16h b3 = *(const v16h*)(wr + 3 * wstep + kk);
        acc0 = __builtin_amdgcn_wmma_f32_16x16x32_f16(false, a, false, b0, (short)0, acc0, false, false);
        acc1 = __builtin_amdgcn_wmma_f32_16x16x32_f16(false, a, false, b1, (short)0, acc1, false, false);
        acc2 = __builtin_amdgcn_wmma_f32_16x16x32_f16(false, a, false, b2, (short)0, acc2, false, false);
        acc3 = __builtin_amdgcn_wmma_f32_16x16x32_f16(false, a, false, b3, (short)0, acc3, false, false);
    }
    int r0 = half * 8;
    v8f accs[4] = {acc0, acc1, acc2, acc3};
    #pragma unroll
    for (int t = 0; t < 4; t++) {
        int col = nbase + t * 16 + lm;
        if (mode == 0) {
            _Float16* Y = (_Float16*)Yv;
            #pragma unroll
            for (int r = 0; r < 8; r++)
                Y[(size_t)(mt * 16 + r0 + r) * C_DIM + col] = (_Float16)accs[t][r];
        } else if (mode == 1) {
            _Float16* Y = (_Float16*)Yv;                 // [B][C][T2P]
            int tok0 = mt * 16 + r0;
            int b = tok0 / T2P, t0 = tok0 % T2P;
            v8h o;
            #pragma unroll
            for (int r = 0; r < 8; r++) o[r] = (_Float16)accs[t][r];
            *(v8h*)(Y + ((size_t)b * C_DIM + col) * T2P + t0) = o;
        } else {
            float* Y = (float*)Yv;
            float bb = bias[col];
            #pragma unroll
            for (int r = 0; r < 8; r++)
                Y[(size_t)(mt * 16 + r0 + r) * C_DIM + col] = accs[t][r] + bb;
        }
    }
}

// ---------------------------------------------------------------------------
// Attention: one block per (qtile, head, batch). 128 threads = 4 waves.
// S = q k^T * SCALE -> LDS f16 [16][800]; masked softmax; O = P v.
// ---------------------------------------------------------------------------
__global__ __launch_bounds__(128) void attn_kernel(const _Float16* __restrict__ qp,
                                                   const _Float16* __restrict__ kp,
                                                   const _Float16* __restrict__ vT,
                                                   _Float16* __restrict__ aout) {
    __shared__ _Float16 sc[16 * T2P];   // 25.6 KB
    int lane = threadIdx.x & 31, wave = threadIdx.x >> 5;
    int qt = blockIdx.x, head = blockIdx.y, b = blockIdx.z;
    int half = lane >> 4, lm = lane & 15;
    int c0 = half * 16;

    // --- q A-fragments (16x64 split into two 16x32 frags), kept in VGPRs ---
    const _Float16* qrow = qp + ((size_t)(b * T1 + qt * 16 + lm)) * C_DIM + head * HD;
    v8h l0 = *(const v8h*)(qrow + c0);
    v8h h0 = *(const v8h*)(qrow + c0 + 8);
    v16h a0 = __builtin_shufflevector(l0, h0, 0,1,2,3,4,5,6,7,8,9,10,11,12,13,14,15);
    v8h l1 = *(const v8h*)(qrow + 32 + c0);
    v8h h1 = *(const v8h*)(qrow + 32 + c0 + 8);
    v16h a1 = __builtin_shufflevector(l1, h1, 0,1,2,3,4,5,6,7,8,9,10,11,12,13,14,15);

    // --- S tiles: 50 tiles of 16 kv cols, round-robin over 4 waves ---
    for (int kt = wave; kt < T2P / 16; kt += 4) {
        int tok = kt * 16 + lm;   // B-frag: lane = kv column
        const _Float16* krow = kp + ((size_t)(b * T2P + tok)) * C_DIM + head * HD + half * 16;
        v16h b0 = *(const v16h*)(krow);
        v16h b1 = *(const v16h*)(krow + 32);
        v8f s = {};
        s = __builtin_amdgcn_wmma_f32_16x16x32_f16(false, a0, false, b0, (short)0, s, false, false);
        s = __builtin_amdgcn_wmma_f32_16x16x32_f16(false, a1, false, b1, (short)0, s, false, false);
        int col = kt * 16 + lm, r0 = half * 8;
        #pragma unroll
        for (int r = 0; r < 8; r++)
            sc[(r0 + r) * T2P + col] = (_Float16)(s[r] * SCALE);
    }
    __syncthreads();

    // --- mask padded kv columns [784,800) ---
    for (int i = threadIdx.x; i < 16 * 16; i += 128) {
        int r = i >> 4, cc = T2 + (i & 15);
        sc[r * T2P + cc] = (_Float16)(-30000.0f);
    }
    __syncthreads();

    // --- softmax: 8 threads per row, shuffle reduce within 8-lane groups ---
    {
        int row = threadIdx.x >> 3, j = threadIdx.x & 7;
        _Float16* srow = sc + row * T2P;
        float mx = -3.0e38f;
        for (int cI = j; cI < T2P; cI += 8) mx = fmaxf(mx, (float)srow[cI]);
        mx = fmaxf(mx, __shfl_xor(mx, 1, 32));
        mx = fmaxf(mx, __shfl_xor(mx, 2, 32));
        mx = fmaxf(mx, __shfl_xor(mx, 4, 32));
        float sum = 0.f;
        for (int cI = j; cI < T2P; cI += 8) {
            float p = __expf((float)srow[cI] - mx);
            sum += p;
            srow[cI] = (_Float16)p;
        }
        sum += __shfl_xor(sum, 1, 32);
        sum += __shfl_xor(sum, 2, 32);
        sum += __shfl_xor(sum, 4, 32);
        float inv = 1.0f / sum;
        for (int cI = j; cI < T2P; cI += 8)
            srow[cI] = (_Float16)((float)srow[cI] * inv);
    }
    __syncthreads();

    // --- O = P @ v : wave w computes d-tile w (16 cols of head dim) ---
    v8f o = {};
    const _Float16* vrow = vT + ((size_t)b * C_DIM + head * HD + wave * 16 + lm) * T2P + half * 16;
    const _Float16* prow = sc + lm * T2P;     // A-frag row = lane&15
    for (int kc = 0; kc < T2P; kc += 32) {
        v8h pl = *(const v8h*)(prow + kc + c0);
        v8h ph = *(const v8h*)(prow + kc + c0 + 8);
        v16h pa = __builtin_shufflevector(pl, ph, 0,1,2,3,4,5,6,7,8,9,10,11,12,13,14,15);
        v16h vb = *(const v16h*)(vrow + kc);
        o = __builtin_amdgcn_wmma_f32_16x16x32_f16(false, pa, false, vb, (short)0, o, false, false);
    }
    int r0 = half * 8;
    int ocol = head * HD + wave * 16 + lm;
    #pragma unroll
    for (int r = 0; r < 8; r++)
        aout[((size_t)(b * T1 + qt * 16 + r0 + r)) * C_DIM + ocol] = (_Float16)o[r];
}

// ---------------------------------------------------------------------------
extern "C" void kernel_launch(void* const* d_in, const int* in_sizes, int n_in,
                              void* d_out, int out_size, void* d_ws, size_t ws_size,
                              hipStream_t stream) {
    const float* x      = (const float*)d_in[0];
    // d_in[1]=h, d_in[2]=w (scalars: H=W=56, hardcoded)
    const float* conv_q = (const float*)d_in[3];
    const float* conv_k = (const float*)d_in[4];
    const float* conv_v = (const float*)d_in[5];
    const float* bn_q   = (const float*)d_in[6];
    const float* bn_k   = (const float*)d_in[7];
    const float* bn_v   = (const float*)d_in[8];
    const float* Wq     = (const float*)d_in[9];
    const float* Wk     = (const float*)d_in[10];
    const float* Wv     = (const float*)d_in[11];
    const float* Wproj  = (const float*)d_in[12];
    const float* bproj  = (const float*)d_in[13];

    // workspace layout (bytes; every size is a multiple of 256)
    const size_t szW  = (size_t)4 * C_DIM * C_DIM * 2;       // 1,179,648
    const size_t szQ  = (size_t)B_SZ * T1 * C_DIM * 2;       // 19,267,584
    const size_t szKV = (size_t)B_SZ * T2P * C_DIM * 2;      //  4,915,200
    char* ws = (char*)d_ws;
    _Float16* wt    = (_Float16*)ws;                         // 4x [384][384] f16^T
    _Float16* qconv = (_Float16*)(ws + szW);                 // [B][T1][C]
    _Float16* kconv = (_Float16*)(ws + szW + szQ);           // [B][T2P][C]
    _Float16* vconv = (_Float16*)(ws + szW + szQ + szKV);    // [B][T2P][C]
    _Float16* qproj = (_Float16*)(ws + szW + szQ + 2 * szKV);       // [B][T1][C]
    _Float16* kproj = (_Float16*)(ws + szW + 2 * szQ + 2 * szKV);   // [B][T2P][C]
    _Float16* vTr   = kconv;   // reuse kconv region after k-GEMM: [B][C][T2P]
    _Float16* aout  = qconv;   // reuse qconv region after q-GEMM: [B][T1][C]

    const int WMAT = C_DIM * C_DIM;

    cvt_weights<<<(4 * WMAT) / 256, 256, 0, stream>>>(Wq, Wk, Wv, Wproj, wt);
    conv_bn_q<<<(B_SZ * T1 * C_DIM) / 256, 256, 0, stream>>>(x, conv_q, bn_q, qconv);
    conv_bn_kv<<<(B_SZ * T2P * C_DIM) / 256, 256, 0, stream>>>(x, conv_k, conv_v,
                                                               bn_k, bn_v, kconv, vconv);
    // q/k/v projections (M multiples of 64: 25088 and 6400)
    gemm_wmma<<<dim3((B_SZ * T1) / 64, C_DIM / 64), 128, 0, stream>>>(
        qconv, wt + 0 * WMAT, nullptr, (void*)qproj, 0);
    gemm_wmma<<<dim3((B_SZ * T2P) / 64, C_DIM / 64), 128, 0, stream>>>(
        kconv, wt + 1 * WMAT, nullptr, (void*)kproj, 0);
    gemm_wmma<<<dim3((B_SZ * T2P) / 64, C_DIM / 64), 128, 0, stream>>>(
        vconv, wt + 2 * WMAT, nullptr, (void*)vTr, 1);
    // attention
    attn_kernel<<<dim3(T1 / 16, HEADS, B_SZ), 128, 0, stream>>>(qproj, kproj, vTr, aout);
    // output projection + bias -> f32 d_out
    gemm_wmma<<<dim3((B_SZ * T1) / 64, C_DIM / 64), 128, 0, stream>>>(
        aout, wt + 3 * WMAT, bproj, d_out, 2);
}
